// Attention_57105885167629
// MI455X (gfx1250) — matrix operand order repaired and checked by reading, not compiled, not run on hardware
//
#include <hip/hip_runtime.h>

#define DIM    4096
#define SEQ    2048
#define BSZ    16
#define NHEADS 32
#define NKV    8
#define NREP   (NHEADS / NKV)
#define HDIM   128

typedef __attribute__((ext_vector_type(16))) __bf16        v16bf;
typedef __attribute__((ext_vector_type(8)))  float         v8f;
typedef __attribute__((ext_vector_type(8)))  unsigned int  u32x8;
typedef __attribute__((ext_vector_type(4)))  unsigned int  u32x4;
typedef __attribute__((ext_vector_type(8)))  int           i32x8;
typedef __attribute__((ext_vector_type(4)))  int           i32x4;

// round-half-up f32 -> bf16 (2 VALU ops per element)
__device__ __forceinline__ unsigned short f2bf1(float f) {
    return (unsigned short)((__float_as_uint(f) + 0x8000u) >> 16);
}
// pack two f32 into one dword of bf16 pairs: lo in [15:0], hi in [31:16]
__device__ __forceinline__ unsigned int pk2bf(float lo, float hi) {
    unsigned int a = (__float_as_uint(lo) + 0x8000u) >> 16;
    unsigned int b = (__float_as_uint(hi) + 0x8000u) & 0xFFFF0000u;
    return a | b;
}

// C[z][m,n] = scale * sum_k A[z][m,k] * (transB ? B[z][n,k] : B[z][k,n])
// One wave (32 threads) per block; block computes a (MT*16) x 16 tile of C.
// z = blockIdx.z picks batch slices via strideA/strideB/strideC; B's batch
// index is z / bdivB (GQA: 4 query heads share one K/V head's weights).
template <int MT>
__global__ void __launch_bounds__(32)
wmma_gemm_bf16(const float* __restrict__ A, long lda, long strideA,
               const float* __restrict__ B, long ldb, long strideB, int bdivB,
               float* __restrict__ C, long ldc, long strideC,
               int K, int transB, float scale)
{
    __shared__ unsigned int As32[MT * 16 * 16]; // MT*16 rows x 32 bf16 (row-major)
    __shared__ unsigned int Bs32[16 * 16];      // 16 n-rows  x 32 bf16 (n-major)
    unsigned short* Bs = (unsigned short*)Bs32;

    const int  lane = threadIdx.x;
    const int  z    = blockIdx.z;
    const long n0   = (long)blockIdx.x * 16;
    const long m0   = (long)blockIdx.y * (MT * 16);

    const float* Az = A + (long)z * strideA;
    const float* Bz = B + (long)(z / bdivB) * strideB;
    float*       Cz = C + (long)z * strideC;

    const v8f vzero = {0.f, 0.f, 0.f, 0.f, 0.f, 0.f, 0.f, 0.f};
    v8f acc[MT];
#pragma unroll
    for (int t = 0; t < MT; ++t) acc[t] = vzero;

    const bool hi   = (lane >= 16);
    const int  ml   = lane & 15;
    // float4-tile coordinates: 8 lanes cover one 32-float row
    const int  arow = lane >> 3;         // row sub-index within group of 4
    const int  acol = (lane & 7) * 4;    // col in {0,4,...,28}
    // !transB: 4 lanes cover one 16-float row
    const int  brow = lane >> 2;
    const int  bcol = (lane & 3) * 4;

    for (int kk = 0; kk < K; kk += 32) {
        // ---- stage A tile: (MT*16) x 32 f32 -> bf16, global_load_b128 each
#pragma unroll
        for (int it = 0; it < MT * 4; ++it) {
            int row = it * 4 + arow;
            float4 f = *(const float4*)&Az[(m0 + row) * lda + (kk + acol)];
            uint2 p; p.x = pk2bf(f.x, f.y); p.y = pk2bf(f.z, f.w);
            *(uint2*)&As32[row * 16 + (acol >> 1)] = p;   // ds_store_b64
        }

        // ---- stage B tile into n-major LDS layout Bs[n][k], 16 x 32
        if (transB) {
#pragma unroll
            for (int it = 0; it < 4; ++it) {
                int row = it * 4 + arow;                  // n row
                float4 f = *(const float4*)&Bz[(n0 + row) * ldb + (kk + acol)];
                uint2 p; p.x = pk2bf(f.x, f.y); p.y = pk2bf(f.z, f.w);
                *(uint2*)&Bs32[row * 16 + (acol >> 1)] = p;
            }
        } else {
#pragma unroll
            for (int it = 0; it < 4; ++it) {
                int krow = it * 8 + brow;                 // k row in global
                float4 f = *(const float4*)&Bz[(long)(kk + krow) * ldb + n0 + bcol];
                Bs[(bcol + 0) * 32 + krow] = f2bf1(f.x); // transpose into n-major
                Bs[(bcol + 1) * 32 + krow] = f2bf1(f.y);
                Bs[(bcol + 2) * 32 + krow] = f2bf1(f.z);
                Bs[(bcol + 3) * 32 + krow] = f2bf1(f.w);
            }
        }
        __syncthreads();

        // ---- prefetch next K tile (global_prefetch_b8)
        if (kk + 32 < K) {
            __builtin_prefetch(&Az[(m0 + arow) * lda + (kk + 32 + acol)], 0, 0);
            if (transB) __builtin_prefetch(&Bz[(n0 + arow) * ldb + (kk + 32 + acol)], 0, 0);
            else        __builtin_prefetch(&Bz[(long)(kk + 32 + brow) * ldb + n0 + bcol], 0, 0);
        }

        // ---- B fragment (32x16 bf16): lane holds n = lane&15,
        //      dword v<4 -> k pair 2v (+8 for hi lanes), v>=4 -> k pair 16+...
        u32x8 bv;
#pragma unroll
        for (int v = 0; v < 8; ++v)
            bv[v] = Bs32[ml * 16 + ((v < 4) ? v : (8 + (v - 4))) + (hi ? 4 : 0)];
        v16bf bfrag = __builtin_bit_cast(v16bf, bv);

        // ---- A fragments (16x32 bf16): lane holds m = lane&15,
        //      dword v<4 -> k pair 2v (+8 for hi lanes), v>=4 -> k pair 16+...
#pragma unroll
        for (int t = 0; t < MT; ++t) {
            u32x8 av;
#pragma unroll
            for (int v = 0; v < 8; ++v) {
                int k0 = ((v < 4) ? (2 * v) : (16 + 2 * (v - 4))) + (hi ? 8 : 0);
                av[v] = As32[(t * 16 + ml) * 16 + (k0 >> 1)];
            }
            v16bf afrag = __builtin_bit_cast(v16bf, av);
            acc[t] = __builtin_amdgcn_wmma_f32_16x16x32_bf16(
                false, afrag, false, bfrag, (short)0, acc[t], false, false);
        }
        __syncthreads();
    }

    // ---- store C: lane -> n = lane&15, dword r -> m = r + (hi ? 8 : 0)
#pragma unroll
    for (int t = 0; t < MT; ++t)
#pragma unroll
        for (int r = 0; r < 8; ++r) {
            long m = m0 + t * 16 + r + (hi ? 8 : 0);
            Cz[m * ldc + n0 + ml] = acc[t][r] * scale;
        }
}

// Softmax over each row of [rows x width].
// Row is DMA'd into LDS by the Tensor Data Mover (one TENSOR_LOAD_TO_LDS with a
// 1-D D# descriptor, TENSORcnt) when the builtin is available, else by
// GLOBAL_LOAD_ASYNC_TO_LDS_B128 (ASYNCcnt). Reductions run from LDS; the
// normalized row is written back with GLOBAL_STORE_ASYNC_FROM_LDS_B128.
__global__ void __launch_bounds__(256)
softmax_rows(float* __restrict__ data, int width)
{
    __shared__ __align__(16) float buf[SEQ];
    __shared__ float red[256];
    float* row = data + (long)blockIdx.x * width;
    const int tid = threadIdx.x;

#if __has_builtin(__builtin_amdgcn_tensor_load_to_lds)
    if (tid < 32) {  // wave 0 issues the DMA once (TDM ignores EXEC)
        // ---- D# group 0 (CDNA5 ISA): count=1 descriptor,
        //      lds_addr in [63:32], 57-bit global_addr in [120:64], type=2
        unsigned long long ga = (unsigned long long)(uintptr_t)row;
        u32x4 g0;
        g0[0] = 1u;                                  // count=1, user mode
        g0[1] = (unsigned)(uintptr_t)buf;            // lds_addr (bytes)
        g0[2] = (unsigned)ga;                        // global_addr[31:0]
        g0[3] = (unsigned)((ga >> 32) & 0x1FFFFFFu)  // global_addr[56:32]
              | (2u << 30);                          // type=2 ("image")
        // ---- D# group 1 (§8.4): 1-D tile of `width` f32 elements
        i32x8 g1;
        g1[0] = (int)(2u << 16);                     // wg_mask=0, data_size=2 (4B)
        g1[1] = (int)((unsigned)width << 16);        // tensor_dim0[15:0] in [63:48]
        g1[2] = (int)(((unsigned)width >> 16) | (1u << 16)); // td0[31:16], tensor_dim1=1
        g1[3] = (int)((unsigned)width << 16);        // tile_dim0 = width
        g1[4] = 1;                                   // tile_dim1=1, tile_dim2 unused
        g1[5] = width;                               // tensor_dim0_stride[31:0]
        g1[6] = (int)((unsigned)width << 16);        // tensor_dim1_stride[15:0]
        g1[7] = 0;                                   // tensor_dim1_stride[47:16]
        i32x4 gz = {0, 0, 0, 0};                     // groups 2/3 unused (<=2D)
#if defined(__clang_major__) && __clang_major__ >= 23
        i32x8 gz8 = {0, 0, 0, 0, 0, 0, 0, 0};
        __builtin_amdgcn_tensor_load_to_lds(g0, g1, gz, gz, gz8, 0);
#else
        __builtin_amdgcn_tensor_load_to_lds(g0, g1, gz, gz, 0);
#endif
        __builtin_amdgcn_s_wait_tensorcnt(0);
    }
#else
    for (int i = 4 * tid; i < width; i += 4 * 256) {
        unsigned       ldsoff = (unsigned)(uintptr_t)&buf[i];
        const float*   gaddr  = row + i;
        asm volatile("global_load_async_to_lds_b128 %0, %1, off"
                     :: "v"(ldsoff), "v"(gaddr) : "memory");
    }
    asm volatile("s_wait_asynccnt 0x0" ::: "memory");
#endif
    __syncthreads();

    float m = -INFINITY;
    for (int i = tid; i < width; i += 256) m = fmaxf(m, buf[i]);
    red[tid] = m; __syncthreads();
    for (int s = 128; s > 0; s >>= 1) {
        if (tid < s) red[tid] = fmaxf(red[tid], red[tid + s]);
        __syncthreads();
    }
    m = red[0]; __syncthreads();

    float sum = 0.f;
    for (int i = tid; i < width; i += 256) {
        float e = __expf(buf[i] - m);
        buf[i] = e;
        sum += e;
    }
    red[tid] = sum; __syncthreads();
    for (int s = 128; s > 0; s >>= 1) {
        if (tid < s) red[tid] += red[tid + s];
        __syncthreads();
    }
    float inv = 1.0f / red[0];
    for (int i = tid; i < width; i += 256) buf[i] *= inv;
    __syncthreads();

    // ---- async copy LDS row -> global (ASYNCcnt path)
    for (int i = 4 * tid; i < width; i += 4 * 256) {
        unsigned ldsoff = (unsigned)(uintptr_t)&buf[i];
        float*   gaddr  = row + i;
        asm volatile("global_store_async_from_lds_b128 %0, %1, off"
                     :: "v"(gaddr), "v"(ldsoff) : "memory");
    }
    asm volatile("s_wait_asynccnt 0x0" ::: "memory");
}

extern "C" void kernel_launch(void* const* d_in, const int* in_sizes, int n_in,
                              void* d_out, int out_size, void* d_ws, size_t ws_size,
                              hipStream_t stream)
{
    const float* x_pre = (const float*)d_in[0]; // [16, 2048, 4096]
    const float* wq    = (const float*)d_in[1]; // [4096, 4096]
    const float* wk    = (const float*)d_in[2]; // [1024, 4096]
    const float* wv    = (const float*)d_in[3]; // [1024, 4096]
    const float* wo    = (const float*)d_in[4]; // [4096, 4096]
    float*       out   = (float*)d_out;         // [16, 1, 4096]

    float* ws     = (float*)d_ws;
    float* q      = ws;                           // 16*4096
    float* temp1  = q      + (long)BSZ * DIM;     // 16*32*4096
    float* scores = temp1  + (long)BSZ * NHEADS * DIM;  // 16*32*2048
    float* ctx    = scores + (long)BSZ * NHEADS * SEQ;  // 16*32*4096
    float* outs   = ctx    + (long)BSZ * NHEADS * DIM;  // 16*4096

    const dim3  blk(32);
    const float rsq = 0.08838834764831845f; // 1/sqrt(128)

    // 1) q[b,:] = x_last[b,:] @ wq^T          (M=16, N=4096, K=4096)
    wmma_gemm_bf16<1><<<dim3(DIM / 16, 1, 1), blk, 0, stream>>>(
        x_pre + (long)(SEQ - 1) * DIM, (long)SEQ * DIM, 0,
        wq, DIM, 0, 1,
        q, DIM, 0,
        DIM, /*transB=*/1, 1.0f);

    // 2) temp1[b,h,:] = q[b, h*128:] @ wk_head    (z=h; M=16, N=4096, K=128)
    wmma_gemm_bf16<1><<<dim3(DIM / 16, 1, NHEADS), blk, 0, stream>>>(
        q, DIM, HDIM,
        wk, DIM, (long)HDIM * DIM, NREP,
        temp1, (long)NHEADS * DIM, DIM,
        HDIM, /*transB=*/0, 1.0f);

    // 3) scores[b,h,t] = temp1[b,h,:]·x_pre[b,t,:]/√128  (z=b; M=32, N=2048, K=4096)
    //    MT=2 keeps all 32 head-rows resident -> x_pre streamed exactly once here.
    wmma_gemm_bf16<2><<<dim3(SEQ / 16, 1, BSZ), blk, 0, stream>>>(
        temp1, DIM, (long)NHEADS * DIM,
        x_pre, DIM, (long)SEQ * DIM, 1,
        scores, SEQ, (long)NHEADS * SEQ,
        DIM, /*transB=*/1, rsq);

    // 4) softmax over t for each of the 512 (b,h) rows
    softmax_rows<<<dim3(BSZ * NHEADS), dim3(256), 0, stream>>>(scores, SEQ);

    // 5) ctx[b,h,:] = attn[b,h,:] @ x_pre[b]   (z=b; M=32, N=4096, K=2048)
    //    (attn @ X) @ Wv^T reordering: second and final stream of x_pre.
    wmma_gemm_bf16<2><<<dim3(DIM / 16, 1, BSZ), blk, 0, stream>>>(
        scores, SEQ, (long)NHEADS * SEQ,
        x_pre, DIM, (long)SEQ * DIM, 1,
        ctx, DIM, (long)NHEADS * DIM,
        SEQ, /*transB=*/0, 1.0f);

    // 6) outs[b, h*128+d] = ctx[b,h,:] @ wv_head^T  (z=h; M=16, N=128, K=4096)
    wmma_gemm_bf16<1><<<dim3(HDIM / 16, 1, NHEADS), blk, 0, stream>>>(
        ctx, (long)NHEADS * DIM, DIM,
        wv, DIM, (long)HDIM * DIM, NREP,
        outs, DIM, HDIM,
        DIM, /*transB=*/1, 1.0f);

    // 7) final[b,:] = outs[b,:] @ wo^T         (M=16, N=4096, K=4096)
    wmma_gemm_bf16<1><<<dim3(DIM / 16, 1, 1), blk, 0, stream>>>(
        outs, DIM, 0,
        wo, DIM, 0, 1,
        out, DIM, 0,
        DIM, /*transB=*/1, 1.0f);
}
